// MultiHeadSelfAttentionBlock_20624432956368
// MI455X (gfx1250) — compile-verified, matrix operand order checked by reading, not run on hardware
//
#include <hip/hip_runtime.h>
#include <math.h>

// ============================================================================
// MI455X (gfx1250) transformer block — TDM + WMMA pipeline, compile-only.
//
// ~206 GFLOP/call, <300MB traffic @23.3TB/s -> every matmul is compute bound.
// bf16 in / fp32 accumulate via v_wmma_f32_16x16x32_bf16.
// GEMM: 128x(64|128) block tile, K-tile 32, 8 waves (4x2), wave tile 32x64
//       (8 WMMA per K-step). Tiles DMA'd into double-buffered LDS by the
//       Tensor Data Mover (tensor_load_to_lds, TENSORcnt), with the D#'s
//       pad fields producing 40-element padded LDS rows directly.
// All weights are pre-transposed to [N][K] bf16 so every GEMM B tile is a
// row-major [rows][K] slice (contiguous K) -> one descriptor shape fits all.
// Attention: flash-style online softmax, one wave per 16-query tile.
//
// NOTE: ping-pong LDS buffers are addressed via run-time offset arithmetic
// (smem + cur*STRIDE); constant-initialized arrays of LDS-derived pointers
// produce addrspacecast static initializers that ld.lld rejects.
// ============================================================================

typedef __attribute__((ext_vector_type(16))) __bf16 v16bf;
typedef __attribute__((ext_vector_type(8)))  float  v8f;

#define DEV __device__ __forceinline__

constexpr int Bc = 2, Tc = 2048, Dc = 1024, Hc = 16, HDc = 64, FFc = 4096;
constexpr int BTc = Bc * Tc;

DEV unsigned short f2bf(float f) {                 // round-to-nearest-even
  unsigned u = __float_as_uint(f);
  u += 0x7fffu + ((u >> 16) & 1u);
  return (unsigned short)(u >> 16);
}
DEV float bf2f(unsigned short h) { return __uint_as_float(((unsigned)h) << 16); }

// K index held in element pair (2v,2v+1) of a 16-bit A/B fragment (ISA 7.12.2)
DEV int kmap(int vi, int hf) { return 2 * (vi & 3) + 16 * (vi >> 2) + 8 * hf; }

union Frag { unsigned u[8]; v16bf v; };

DEV v8f wmma_bf16(v16bf a, v16bf b, v8f c) {
  return __builtin_amdgcn_wmma_f32_16x16x32_bf16(false, a, false, b, (short)0, c,
                                                 false, false);
}

DEV float block_sum_256(float v, float* red) {     // 256 threads = 8 waves
  #pragma unroll
  for (int off = 16; off > 0; off >>= 1) v += __shfl_xor(v, off, 32);
  if ((threadIdx.x & 31) == 0) red[threadIdx.x >> 5] = v;
  __syncthreads();
  float t = (threadIdx.x < 8) ? red[threadIdx.x] : 0.0f;
  #pragma unroll
  for (int off = 4; off > 0; off >>= 1) t += __shfl_xor(t, off, 32);
  if (threadIdx.x == 0) red[0] = t;
  __syncthreads();
  return red[0];
}

// ---------------------------------------------------------------------------
// Tensor Data Mover: async 2D tile load global->LDS (ISA 08_async_tensor §7/8)
// Loads a [rows x 32] bf16 tile whose rows are K elements apart, padding the
// LDS destination to 40-element rows (pad 4 dwords every 16 dwords).
// ---------------------------------------------------------------------------
#if __has_builtin(__builtin_amdgcn_tensor_load_to_lds) && \
    __has_builtin(__builtin_amdgcn_s_wait_tensorcnt)
#define HAS_TDM 1
#else
#define HAS_TDM 0
#endif

#if HAS_TDM
typedef __attribute__((ext_vector_type(4))) unsigned v4u_t;
typedef __attribute__((ext_vector_type(4))) int      v4i_t;
typedef __attribute__((ext_vector_type(8))) int      v8i_t;

DEV void tdm_load_2d(unsigned lds_byte_off, const unsigned short* gsrc,
                     int rows, int K) {
  const unsigned long long ga = (unsigned long long)(uintptr_t)gsrc;
  v4u_t g0;
  g0[0] = 1u;                                   // count=1, user descriptor
  g0[1] = lds_byte_off;                         // lds_addr
  g0[2] = (unsigned)(ga & 0xffffffffu);         // global_addr[31:0]
  g0[3] = (unsigned)((ga >> 32) & 0x1ffffffu)   // global_addr[56:32]
          | (2u << 30);                         // type = 2 ("image")
  v8i_t g1;
  g1[0] = (int)((1u << 16)                      // data_size = 2 bytes
                | (1u << 20)                    // pad_enable
                | (3u << 22)                    // pad_interval: 16 dwords (64B)
                | (3u << 25));                  // pad_amount: 4 dwords (16B)
  g1[1] = (int)(((unsigned)K & 0xffffu) << 16);               // tensor_dim0 lo
  g1[2] = (int)((((unsigned)K >> 16) & 0xffffu) |             // tensor_dim0 hi
                (((unsigned)rows & 0xffffu) << 16));          // tensor_dim1 lo
  g1[3] = (int)((((unsigned)rows >> 16) & 0xffffu) |          // tensor_dim1 hi
                (32u << 16));                                 // tile_dim0 = 32
  g1[4] = (int)((unsigned)rows & 0xffffu);                    // tile_dim1
  g1[5] = (int)K;                                             // dim0_stride lo
  g1[6] = 0;                                                  // stride hi / d1s
  g1[7] = 0;
  const v4i_t z4 = {0, 0, 0, 0};
#if defined(__clang_major__) && __clang_major__ >= 23
  const v8i_t z8 = {0, 0, 0, 0, 0, 0, 0, 0};
  __builtin_amdgcn_tensor_load_to_lds(g0, g1, z4, z4, z8, 0);
#else
  __builtin_amdgcn_tensor_load_to_lds(g0, g1, z4, z4, 0);
#endif
}

DEV unsigned lds_off_of(const void* p) { return (unsigned)(uintptr_t)p; }
#endif  // HAS_TDM

// ---------------------------------------------------------------------------
// fp32 -> bf16 conversions (weights, once per launch)
// ---------------------------------------------------------------------------
__global__ __launch_bounds__(256) void cvt_bf16_k(const float* __restrict__ in,
                                                  unsigned short* __restrict__ out,
                                                  int n) {
  const int i = blockIdx.x * 256 + threadIdx.x;
  if (i < n) out[i] = f2bf(in[i]);
}

// transpose + convert: in [R][C] fp32 (slab z) -> out [C][R] bf16
__global__ __launch_bounds__(256) void tcvt_bf16_k(const float* __restrict__ in,
                                                   unsigned short* __restrict__ out,
                                                   int R, int C) {
  __shared__ float tile[32][33];
  const size_t slab = (size_t)blockIdx.z * R * C;
  const int c0 = blockIdx.x * 32, r0 = blockIdx.y * 32;
  const int tx = threadIdx.x & 31, ty = threadIdx.x >> 5;  // ty: 0..7
  #pragma unroll
  for (int i = ty; i < 32; i += 8)
    tile[i][tx] = in[slab + (size_t)(r0 + i) * C + c0 + tx];
  __syncthreads();
  #pragma unroll
  for (int i = ty; i < 32; i += 8)
    out[slab + (size_t)(c0 + i) * R + r0 + tx] = f2bf(tile[tx][i]);
}

// ---------------------------------------------------------------------------
// RMSNorm -> bf16 (one block per row of D=1024)
// ---------------------------------------------------------------------------
__global__ __launch_bounds__(256) void rmsnorm_k(const float* __restrict__ x,
                                                 const float* __restrict__ w,
                                                 unsigned short* __restrict__ out) {
  __shared__ float red[8];
  const size_t base = (size_t)blockIdx.x * Dc;
  float ss = 0.f;
  for (int i = threadIdx.x; i < Dc; i += 256) { float t = x[base + i]; ss += t * t; }
  ss = block_sum_256(ss, red);
  const float inv = rsqrtf(ss * (1.0f / Dc) + 1e-6f);
  for (int i = threadIdx.x; i < Dc; i += 256)
    out[base + i] = f2bf(w[i] * x[base + i] * inv);
}

// residual add + RMSNorm: x1 = x + add (fp32), xf_bf = rmsnorm(x1)*w (bf16)
__global__ __launch_bounds__(256) void add_rmsnorm_k(const float* __restrict__ x,
                                                     const float* __restrict__ add,
                                                     const float* __restrict__ w,
                                                     float* __restrict__ xsum,
                                                     unsigned short* __restrict__ out) {
  __shared__ float red[8];
  const size_t base = (size_t)blockIdx.x * Dc;
  float ss = 0.f;
  for (int i = threadIdx.x; i < Dc; i += 256) {
    float t = x[base + i] + add[base + i];
    xsum[base + i] = t;
    ss += t * t;
  }
  ss = block_sum_256(ss, red);
  const float inv = rsqrtf(ss * (1.0f / Dc) + 1e-6f);
  for (int i = threadIdx.x; i < Dc; i += 256)
    out[base + i] = f2bf(w[i] * xsum[base + i] * inv);
}

// ---------------------------------------------------------------------------
// RoPE on bf16 q,k buffers [B,H,T,HD]; one (even,odd) pair per thread.
// ---------------------------------------------------------------------------
__global__ __launch_bounds__(256) void rope_k(unsigned short* __restrict__ q,
                                              unsigned short* __restrict__ k,
                                              const float* __restrict__ cs,
                                              const float* __restrict__ sn) {
  const size_t idx = (size_t)blockIdx.x * 256 + threadIdx.x;
  const int e2 = (int)(idx & (HDc / 2 - 1));
  const int t  = (int)((idx >> 5) & (Tc - 1));
  const float c = cs[t * (HDc / 2) + e2];
  const float s = sn[t * (HDc / 2) + e2];
  const size_t p = idx * 2;
  {
    const float xr = bf2f(q[p]), xi = bf2f(q[p + 1]);
    q[p]     = f2bf(xr * c - xi * s);
    q[p + 1] = f2bf(xr * s + xi * c);
  }
  {
    const float xr = bf2f(k[p]), xi = bf2f(k[p + 1]);
    k[p]     = f2bf(xr * c - xi * s);
    k[p + 1] = f2bf(xr * s + xi * c);
  }
}

// ---------------------------------------------------------------------------
// Tiled WMMA GEMM with TDM double-buffered LDS staging.
//   C[M,N] = A[M,K] x B^T where B is stored [N][K] (weights pre-transposed).
//   Block tile 128 x TBN (TBN=64|128), K-tile 32, 8 waves as 4x2.
//   Wave tile 32 x TBN/2: 2 A-frags x (TBN/32) B-frags -> up to 8 WMMA/K-step.
//   Epilogues: SILU, MULBF (x bf16 tensor), RESID (+fp32 tensor), BIAS.
//   OUT_HEAD scatters [M,N]->[B,H,T,HD] with head = blockIdx.z (B offset too).
// ---------------------------------------------------------------------------
enum { OUT_F32 = 0, OUT_BF = 1, OUT_HEAD = 2 };
constexpr int LDST = 40;  // padded LDS row stride (elements); 80B = 16B multiple

template <int TBN, int OUTMODE, bool RESID, bool BIAS, bool SILU, bool MULBF>
__global__ __launch_bounds__(256) void gemm_wmma_k(
    const unsigned short* __restrict__ A, const unsigned short* __restrict__ Bm,
    void* __restrict__ Out, const float* __restrict__ Resid,
    const float* __restrict__ Bias, const unsigned short* __restrict__ MulIn,
    int M, int N, int K) {
  constexpr int TBM = 128;
  constexpr int NFR = TBN / 32;          // B fragments per wave
  constexpr int ASZ = TBM * LDST;        // elements per A buffer
  constexpr int BSZ = TBN * LDST;
  constexpr int BUF = ASZ + BSZ;         // one ping-pong buffer (A then B)
  __shared__ __align__(16) unsigned short smem[2 * BUF];

  const int tid = threadIdx.x;
  const int lane = tid & 31, wv = tid >> 5;
  const int wm = wv >> 1, wn = wv & 1;
  const int hf = lane >> 4, l15 = lane & 15;
  const int bM = blockIdx.x * TBM, bN = blockIdx.y * TBN;

  const unsigned short* Bp = Bm;
  if constexpr (OUTMODE == OUT_HEAD) Bp += (size_t)blockIdx.z * N * K;

  v8f acc[2][NFR] = {};

  const int nk = K / 32;
#if HAS_TDM
  if (wv == 0) {  // wave 0 drives the Tensor Data Mover
    tdm_load_2d(lds_off_of(smem), A + (size_t)bM * K, TBM, K);
    tdm_load_2d(lds_off_of(smem + ASZ), Bp + (size_t)bN * K, TBN, K);
    __builtin_amdgcn_s_wait_tensorcnt(0);
  }
  __syncthreads();
#endif
  for (int t = 0; t < nk; ++t) {
    const int cur = t & 1;
    unsigned short* Acur = smem + cur * BUF;        // runtime offset arithmetic
    unsigned short* Bcur = Acur + ASZ;              // (no LDS-pointer constants)
#if HAS_TDM
    if (wv == 0 && t + 1 < nk) {  // prefetch next K-tile into other buffer
      unsigned short* Anxt = smem + (cur ^ 1) * BUF;
      tdm_load_2d(lds_off_of(Anxt), A + (size_t)bM * K + (t + 1) * 32, TBM, K);
      tdm_load_2d(lds_off_of(Anxt + ASZ), Bp + (size_t)bN * K + (t + 1) * 32, TBN, K);
    }
#else
    {  // fallback: cooperative vector loads, 8 bf16 (one b128) per thread step
      #pragma unroll
      for (int e = tid * 8; e < TBM * 32; e += 256 * 8) {
        const int row = e >> 5, col = e & 31;
        *(uint4*)&Acur[row * LDST + col] =
            *(const uint4*)&A[(size_t)(bM + row) * K + t * 32 + col];
      }
      #pragma unroll
      for (int e = tid * 8; e < TBN * 32; e += 256 * 8) {
        const int row = e >> 5, col = e & 31;
        *(uint4*)&Bcur[row * LDST + col] =
            *(const uint4*)&Bp[(size_t)(bN + row) * K + t * 32 + col];
      }
    }
    __syncthreads();
#endif
    Frag a[2], b[NFR];
    #pragma unroll
    for (int vi = 0; vi < 8; ++vi) {
      const int k = kmap(vi, hf);  // (k,k+1) pair contiguous -> ds_load_b32
      a[0].u[vi] = *(const unsigned*)&Acur[(wm * 32 + l15) * LDST + k];
      a[1].u[vi] = *(const unsigned*)&Acur[(wm * 32 + 16 + l15) * LDST + k];
      #pragma unroll
      for (int j = 0; j < NFR; ++j)
        b[j].u[vi] = *(const unsigned*)&Bcur[(wn * (TBN / 2) + j * 16 + l15) * LDST + k];
    }
    #pragma unroll
    for (int i = 0; i < 2; ++i)
      #pragma unroll
      for (int j = 0; j < NFR; ++j)
        acc[i][j] = wmma_bf16(a[i].v, b[j].v, acc[i][j]);
#if HAS_TDM
    if (wv == 0 && t + 1 < nk) __builtin_amdgcn_s_wait_tensorcnt(0);
#endif
    __syncthreads();
  }

  #pragma unroll
  for (int i = 0; i < 2; ++i) {
    #pragma unroll
    for (int j = 0; j < NFR; ++j) {
      #pragma unroll
      for (int r = 0; r < 8; ++r) {
        const int m = bM + wm * 32 + i * 16 + r + 8 * hf;  // C: VGPR r -> M row
        const int n = bN + wn * (TBN / 2) + j * 16 + l15;  // C: lane -> N col
        float val = acc[i][j][r];
        if constexpr (SILU)  val = val / (1.0f + __expf(-val));
        if constexpr (MULBF) val *= bf2f(MulIn[(size_t)m * N + n]);
        if constexpr (RESID) val += Resid[(size_t)m * N + n];
        if constexpr (BIAS)  val += Bias[n];
        if constexpr (OUTMODE == OUT_F32) {
          ((float*)Out)[(size_t)m * N + n] = val;
        } else if constexpr (OUTMODE == OUT_BF) {
          ((unsigned short*)Out)[(size_t)m * N + n] = f2bf(val);
        } else {  // OUT_HEAD: m = b*T + t -> [b, h=blockIdx.z, t, n]
          const int bb = m / Tc, tt = m % Tc;
          ((unsigned short*)Out)[(((size_t)bb * Hc + blockIdx.z) * Tc + tt) * HDc + n] =
              f2bf(val);
        }
      }
    }
  }
}

// ---------------------------------------------------------------------------
// Flash attention (causal, scale 1/8): one wave per 16-query tile, 32-key
// tiles, online softmax. P transposed through 1KB/wave LDS to become the A
// operand of P@V. No __syncthreads inside the (wave-divergent) key loop.
// ---------------------------------------------------------------------------
__global__ __launch_bounds__(256) void attn_flash_k(
    const unsigned short* __restrict__ qb, const unsigned short* __restrict__ kb,
    const unsigned short* __restrict__ vb, float* __restrict__ attn) {
  __shared__ __align__(4) unsigned short Plds[8][16 * 34];  // 16x32 + pad
  const int bh = blockIdx.x;                 // b*H + h
  const int wv = threadIdx.x >> 5, lane = threadIdx.x & 31;
  const int hf = lane >> 4, l15 = lane & 15;
  const int qt = blockIdx.y * 8 + wv;        // query tile (16 rows)

  const size_t base = (size_t)bh * Tc * HDc;
  const unsigned short* Q  = qb + base;
  const unsigned short* Kp = kb + base;
  const unsigned short* Vp = vb + base;

  Frag qa0, qa1;  // Q rows, HD 0..31 and 32..63
  const int qrow = qt * 16 + l15;
  #pragma unroll
  for (int vi = 0; vi < 8; ++vi) {
    const int k = kmap(vi, hf);
    qa0.u[vi] = *(const unsigned*)&Q[(size_t)qrow * HDc + k];
    qa1.u[vi] = *(const unsigned*)&Q[(size_t)qrow * HDc + 32 + k];
  }

  v8f o[4] = {};
  float m_i[8], l_i[8];
  #pragma unroll
  for (int r = 0; r < 8; ++r) { m_i[r] = -1e30f; l_i[r] = 0.f; }

  unsigned short* Pw = &Plds[wv][0];
  const int smax = qt >> 1;  // last key tile touching the causal triangle
  for (int s = 0; s <= smax; ++s) {
    const int key0 = s * 32;
    Frag k00, k01, k10, k11;  // K^T B-fragments: lane -> key column
    #pragma unroll
    for (int vi = 0; vi < 8; ++vi) {
      const int k = kmap(vi, hf);
      k00.u[vi] = *(const unsigned*)&Kp[(size_t)(key0 + l15) * HDc + k];
      k01.u[vi] = *(const unsigned*)&Kp[(size_t)(key0 + l15) * HDc + 32 + k];
      k10.u[vi] = *(const unsigned*)&Kp[(size_t)(key0 + 16 + l15) * HDc + k];
      k11.u[vi] = *(const unsigned*)&Kp[(size_t)(key0 + 16 + l15) * HDc + 32 + k];
    }
    v8f cl0 = {}, cl1 = {};
    cl0 = wmma_bf16(qa0.v, k00.v, cl0);
    cl0 = wmma_bf16(qa1.v, k01.v, cl0);
    cl1 = wmma_bf16(qa0.v, k10.v, cl1);
    cl1 = wmma_bf16(qa1.v, k11.v, cl1);

    #pragma unroll
    for (int r = 0; r < 8; ++r) {
      const int row = qt * 16 + r + 8 * hf;
      float v0 = cl0[r] * 0.125f + ((key0 + l15)      > row ? -1e9f : 0.f);
      float v1 = cl1[r] * 0.125f + ((key0 + 16 + l15) > row ? -1e9f : 0.f);
      float tmax = fmaxf(v0, v1);
      #pragma unroll
      for (int off = 1; off < 16; off <<= 1)
        tmax = fmaxf(tmax, __shfl_xor(tmax, off, 32));   // stays within half
      const float mnew = fmaxf(m_i[r], tmax);
      const float corr = __expf(m_i[r] - mnew);
      const float p0 = __expf(v0 - mnew);
      const float p1 = __expf(v1 - mnew);
      float rs = p0 + p1;
      #pragma unroll
      for (int off = 1; off < 16; off <<= 1) rs += __shfl_xor(rs, off, 32);
      l_i[r] = l_i[r] * corr + rs;
      m_i[r] = mnew;
      #pragma unroll
      for (int f = 0; f < 4; ++f) o[f][r] *= corr;
      const int prow = r + 8 * hf;
      Pw[prow * 34 + l15]      = f2bf(p0);
      Pw[prow * 34 + 16 + l15] = f2bf(p1);
    }
    __builtin_amdgcn_wave_barrier();  // same-wave LDS is in-order

    Frag pa;  // P as A operand: lane -> query row, K -> key-in-tile
    #pragma unroll
    for (int vi = 0; vi < 8; ++vi) {
      const int k = kmap(vi, hf);
      pa.u[vi] = *(const unsigned*)&Pw[l15 * 34 + k];
    }
    #pragma unroll
    for (int f = 0; f < 4; ++f) {  // V B-fragments: lane -> HD column
      Frag vf;
      #pragma unroll
      for (int vi = 0; vi < 8; ++vi) {
        const int k = kmap(vi, hf);
        const unsigned lo = Vp[(size_t)(key0 + k) * HDc + f * 16 + l15];
        const unsigned hi = Vp[(size_t)(key0 + k + 1) * HDc + f * 16 + l15];
        vf.u[vi] = lo | (hi << 16);
      }
      o[f] = wmma_bf16(pa.v, vf.v, o[f]);
    }
    __builtin_amdgcn_wave_barrier();
  }

  const int bb = bh / Hc, h = bh % Hc;
  #pragma unroll
  for (int f = 0; f < 4; ++f) {
    #pragma unroll
    for (int r = 0; r < 8; ++r) {
      const int trow = qt * 16 + r + 8 * hf;
      attn[((size_t)bb * Tc + trow) * Dc + h * HDc + f * 16 + l15] =
          o[f][r] / l_i[r];
    }
  }
}

// ---------------------------------------------------------------------------
// Workspace layout (bytes); all weight regions hold B^T ([N][K]) bf16.
// ---------------------------------------------------------------------------
constexpr size_t OFF_WQ  = 0;
constexpr size_t OFF_WK  = OFF_WQ  + (size_t)Hc * Dc * HDc * 2;
constexpr size_t OFF_WV  = OFF_WK  + (size_t)Hc * Dc * HDc * 2;
constexpr size_t OFF_PRJ = OFF_WV  + (size_t)Hc * Dc * HDc * 2;
constexpr size_t OFF_FC1 = OFF_PRJ + (size_t)Dc * Dc * 2;
constexpr size_t OFF_FC2 = OFF_FC1 + (size_t)Dc * FFc * 2;
constexpr size_t OFF_FC3 = OFF_FC2 + (size_t)Dc * FFc * 2;
constexpr size_t OFF_XN  = OFF_FC3 + (size_t)FFc * Dc * 2;
constexpr size_t OFF_Q   = OFF_XN  + (size_t)BTc * Dc * 2;
constexpr size_t OFF_K   = OFF_Q   + (size_t)BTc * Dc * 2;
constexpr size_t OFF_V   = OFF_K   + (size_t)BTc * Dc * 2;
constexpr size_t OFF_ATT = OFF_V   + (size_t)BTc * Dc * 2;
constexpr size_t OFF_X1  = OFF_ATT + (size_t)BTc * Dc * 4;
constexpr size_t OFF_XF  = OFF_X1  + (size_t)BTc * Dc * 4;
constexpr size_t OFF_S   = OFF_XF  + (size_t)BTc * Dc * 2;
constexpr size_t OFF_P   = OFF_S   + (size_t)BTc * FFc * 2;
constexpr size_t OFF_X2  = OFF_P   + (size_t)BTc * FFc * 2;  // end ~185 MB

extern "C" void kernel_launch(void* const* d_in, const int* in_sizes, int n_in,
                              void* d_out, int out_size, void* d_ws, size_t ws_size,
                              hipStream_t stream) {
  (void)in_sizes; (void)n_in; (void)out_size; (void)ws_size;
  const float* x     = (const float*)d_in[0];
  const float* cosb  = (const float*)d_in[1];
  const float* sinb  = (const float*)d_in[2];
  // d_in[3]: mask — causal handled analytically
  const float* att_w = (const float*)d_in[4];
  const float* Wq    = (const float*)d_in[5];
  const float* Wk    = (const float*)d_in[6];
  const float* Wv    = (const float*)d_in[7];
  const float* ffn_w = (const float*)d_in[8];
  const float* fc1   = (const float*)d_in[9];
  const float* fc2   = (const float*)d_in[10];
  const float* fc3   = (const float*)d_in[11];
  const float* projw = (const float*)d_in[12];
  const float* projb = (const float*)d_in[13];
  float* out = (float*)d_out;

  char* ws = (char*)d_ws;
  auto U16 = [&](size_t off) { return (unsigned short*)(ws + off); };
  auto F32 = [&](size_t off) { return (float*)(ws + off); };

  // --- weights -> bf16, transposed to [N][K] where needed ---
  // Wq/Wk/Wv: [h][D][HD] -> [h][HD][D]
  tcvt_bf16_k<<<dim3(HDc / 32, Dc / 32, Hc), 256, 0, stream>>>(Wq, U16(OFF_WQ), Dc, HDc);
  tcvt_bf16_k<<<dim3(HDc / 32, Dc / 32, Hc), 256, 0, stream>>>(Wk, U16(OFF_WK), Dc, HDc);
  tcvt_bf16_k<<<dim3(HDc / 32, Dc / 32, Hc), 256, 0, stream>>>(Wv, U16(OFF_WV), Dc, HDc);
  // fc1/fc2: [D][FF] -> [FF][D];  fc3: [FF][D] -> [D][FF]
  tcvt_bf16_k<<<dim3(FFc / 32, Dc / 32, 1), 256, 0, stream>>>(fc1, U16(OFF_FC1), Dc, FFc);
  tcvt_bf16_k<<<dim3(FFc / 32, Dc / 32, 1), 256, 0, stream>>>(fc2, U16(OFF_FC2), Dc, FFc);
  tcvt_bf16_k<<<dim3(Dc / 32, FFc / 32, 1), 256, 0, stream>>>(fc3, U16(OFF_FC3), FFc, Dc);
  // proj_w is already [N][K] for x @ W^T
  cvt_bf16_k<<<(Dc * Dc + 255) / 256, 256, 0, stream>>>(projw, U16(OFF_PRJ), Dc * Dc);

  rmsnorm_k<<<BTc, 256, 0, stream>>>(x, att_w, U16(OFF_XN));

  // Q,K,V: per-head GEMM, M=4096, N=64, K=1024; grid.z = head
  dim3 gQKV(BTc / 128, 1, Hc);
  gemm_wmma_k<64, OUT_HEAD, false, false, false, false><<<gQKV, 256, 0, stream>>>(
      U16(OFF_XN), U16(OFF_WQ), U16(OFF_Q), nullptr, nullptr, nullptr, BTc, HDc, Dc);
  gemm_wmma_k<64, OUT_HEAD, false, false, false, false><<<gQKV, 256, 0, stream>>>(
      U16(OFF_XN), U16(OFF_WK), U16(OFF_K), nullptr, nullptr, nullptr, BTc, HDc, Dc);
  gemm_wmma_k<64, OUT_HEAD, false, false, false, false><<<gQKV, 256, 0, stream>>>(
      U16(OFF_XN), U16(OFF_WV), U16(OFF_V), nullptr, nullptr, nullptr, BTc, HDc, Dc);

  const size_t npair = (size_t)Bc * Hc * Tc * (HDc / 2);
  rope_k<<<(unsigned)(npair / 256), 256, 0, stream>>>(U16(OFF_Q), U16(OFF_K),
                                                      cosb, sinb);

  attn_flash_k<<<dim3(Bc * Hc, Tc / (16 * 8)), 256, 0, stream>>>(
      U16(OFF_Q), U16(OFF_K), U16(OFF_V), F32(OFF_ATT));

  add_rmsnorm_k<<<BTc, 256, 0, stream>>>(x, F32(OFF_ATT), ffn_w, F32(OFF_X1),
                                         U16(OFF_XF));

  // FFN: s = bf16(silu(xf @ fc1)); p = bf16((xf @ fc2) * s); x2 = x1 + p @ fc3
  dim3 gFF(BTc / 128, FFc / 128);
  gemm_wmma_k<128, OUT_BF, false, false, true, false><<<gFF, 256, 0, stream>>>(
      U16(OFF_XF), U16(OFF_FC1), U16(OFF_S), nullptr, nullptr, nullptr, BTc, FFc, Dc);
  gemm_wmma_k<128, OUT_BF, false, false, false, true><<<gFF, 256, 0, stream>>>(
      U16(OFF_XF), U16(OFF_FC2), U16(OFF_P), nullptr, nullptr, U16(OFF_S), BTc, FFc, Dc);
  dim3 gD(BTc / 128, Dc / 128);
  gemm_wmma_k<128, OUT_BF, true, false, false, false><<<gD, 256, 0, stream>>>(
      U16(OFF_P), U16(OFF_FC3), U16(OFF_X2), F32(OFF_X1), nullptr, nullptr, BTc, Dc, FFc);

  // out = x2 @ proj_w^T + proj_b
  gemm_wmma_k<128, OUT_F32, false, true, false, false><<<gD, 256, 0, stream>>>(
      U16(OFF_X2), U16(OFF_PRJ), out, nullptr, projb, nullptr, BTc, Dc, Dc);
}
